// SapBERTembeddedSOM_8916352107120
// MI455X (gfx1250) — compile-verified, matrix operand order checked by reading, not run on hardware
//
#include <hip/hip_runtime.h>
#include <hip/hip_bf16.h>

#define COMPS 16384
#define DIM   768
#define BS    1024

typedef __bf16 bf16_t;
typedef __attribute__((ext_vector_type(16))) __bf16 v16bf;
typedef __attribute__((ext_vector_type(8)))  __bf16 v8bf;
typedef __attribute__((ext_vector_type(8)))  float  v8f;

static __device__ __forceinline__ bf16_t f2bf(float f) {
  unsigned u = __float_as_uint(f);
  unsigned r = u + 0x7FFFu + ((u >> 16) & 1u);   // round-to-nearest-even
  unsigned short h = (unsigned short)(r >> 16);
  bf16_t b;
  __builtin_memcpy(&b, &h, sizeof(h));
  return b;
}

static __device__ __forceinline__ v16bf ldfrag(const bf16_t* p) {
  v8bf lo = *(const v8bf*)(p);
  v8bf hi = *(const v8bf*)(p + 16);
  return __builtin_shufflevector(lo, hi, 0,1,2,3,4,5,6,7,8,9,10,11,12,13,14,15);
}

// ---------------- init ----------------
__global__ void k_init(unsigned long long* __restrict__ keys, float* __restrict__ out_loss) {
  int i = blockIdx.x * blockDim.x + threadIdx.x;
  if (i < BS) keys[i] = ~0ULL;
  if (i == 0) out_loss[0] = 0.f;
}

// -------- convert X -> bf16 (row-major + transposed), rowwise |x|^2 --------
__global__ __launch_bounds__(256) void k_conv_x(const float* __restrict__ x,
                                                float* __restrict__ x2,
                                                bf16_t* __restrict__ xbf,
                                                bf16_t* __restrict__ xtbf) {
  __shared__ float red[256];
  int b = blockIdx.x, t = threadIdx.x;
  float s = 0.f;
  for (int i = 0; i < DIM / 256; ++i) {
    int d = t + i * 256;
    float v = x[(size_t)b * DIM + d];
    s += v * v;
    bf16_t h = f2bf(v);
    xbf[(size_t)b * DIM + d] = h;
    xtbf[(size_t)d * BS + b] = h;     // Xᵀ for GEMM2's B operand
  }
  red[t] = s; __syncthreads();
  for (int off = 128; off > 0; off >>= 1) { if (t < off) red[t] += red[t + off]; __syncthreads(); }
  if (t == 0) x2[b] = red[0];
}

// -------- convert W -> bf16, rowwise |w|^2 --------
__global__ __launch_bounds__(256) void k_conv_w(const float* __restrict__ w,
                                                float* __restrict__ w2,
                                                bf16_t* __restrict__ wbf) {
  __shared__ float red[256];
  int c = blockIdx.x, t = threadIdx.x;
  float s = 0.f;
  for (int i = 0; i < DIM / 256; ++i) {
    int d = t + i * 256;
    float v = w[(size_t)c * DIM + d];
    s += v * v;
    wbf[(size_t)c * DIM + d] = f2bf(v);
  }
  red[t] = s; __syncthreads();
  for (int off = 128; off > 0; off >>= 1) { if (t < off) red[t] += red[t + off]; __syncthreads(); }
  if (t == 0) w2[c] = red[0];
}

// -------- GEMM1: dist = sqrt(x2 + w2 - 2*X@Wᵀ), fused argmin via u64 atomic-min --------
// Wave computes 64(M) x 32(N) via 4x2 register blocking with a single base pointer
// per matrix (constant instruction offsets -> global_load_b128, not flat).
__global__ __launch_bounds__(256) void k_gemm1_bmu(const bf16_t* __restrict__ xbf,
                                                   const bf16_t* __restrict__ wbf,
                                                   const float*  __restrict__ x2,
                                                   const float*  __restrict__ w2,
                                                   unsigned long long* __restrict__ keys) {
  const int lane  = threadIdx.x & 31;
  const int wv    = threadIdx.x >> 5;
  const int half  = lane >> 4;
  const int tileM = blockIdx.x * 128 + (wv >> 2) * 64;   // batch rows
  const int tileN = blockIdx.y * 128 + (wv & 3) * 32;    // neurons
  const int koff  = half << 3;

  const bf16_t* pA = xbf + (size_t)(tileM + (lane & 15)) * DIM + koff;
  const bf16_t* pB = wbf + (size_t)(tileN + (lane & 15)) * DIM + koff;

  v8f acc[4][2];
#pragma unroll
  for (int i = 0; i < 4; ++i)
#pragma unroll
    for (int j = 0; j < 2; ++j) acc[i][j] = (v8f){};

  for (int k = 0; k < DIM; k += 32) {
    v16bf af[4], bfr[2];
#pragma unroll
    for (int i = 0; i < 4; ++i) af[i] = ldfrag(pA + (size_t)(i * 16) * DIM);
#pragma unroll
    for (int j = 0; j < 2; ++j) bfr[j] = ldfrag(pB + (size_t)(j * 16) * DIM);
    __builtin_prefetch(pA + 32, 0, 1);
    __builtin_prefetch(pB + 32, 0, 1);
#pragma unroll
    for (int i = 0; i < 4; ++i)
#pragma unroll
      for (int j = 0; j < 2; ++j)
        acc[i][j] = __builtin_amdgcn_wmma_f32_16x16x32_bf16(
            false, af[i], false, bfr[j], (short)0, acc[i][j], false, false);
    pA += 32; pB += 32;
  }

  // epilogue: distance + argmin key, fold the two N-tiles before lane reduction
  float w2n[2];
  int   ncol[2];
#pragma unroll
  for (int j = 0; j < 2; ++j) {
    ncol[j] = tileN + j * 16 + (lane & 15);
    w2n[j]  = w2[ncol[j]];
  }
#pragma unroll
  for (int i = 0; i < 4; ++i) {
#pragma unroll
    for (int r = 0; r < 8; ++r) {
      int m = tileM + i * 16 + r + (half << 3);
      float x2m = x2[m];
      unsigned long long key = ~0ULL;
#pragma unroll
      for (int j = 0; j < 2; ++j) {
        float dist = __builtin_amdgcn_sqrtf(fmaxf(x2m + w2n[j] - 2.f * acc[i][j][r], 0.f));
        unsigned long long kj =
            ((unsigned long long)__float_as_uint(dist) << 32) | (unsigned)ncol[j];
        key = (kj < key) ? kj : key;
      }
      // min over the 16 lanes sharing row m (dist>=0 -> bits monotonic; ties -> lowest index)
      for (int mask = 1; mask < 16; mask <<= 1) {
        unsigned lo = __shfl_xor((unsigned)key, mask, 32);
        unsigned hi = __shfl_xor((unsigned)(key >> 32), mask, 32);
        unsigned long long o = ((unsigned long long)hi << 32) | lo;
        key = (o < key) ? o : key;
      }
      if ((lane & 15) == 0) atomicMin(&keys[m], key);
    }
  }
}

// -------- decode BMU keys, accumulate loss --------
__global__ void k_bmu(const unsigned long long* __restrict__ keys,
                      const int* __restrict__ loc,
                      float* __restrict__ bmuI, float* __restrict__ bmuJ,
                      float* __restrict__ out_loss) {
  int b = blockIdx.x * blockDim.x + threadIdx.x;
  if (b >= BS) return;
  unsigned long long key = keys[b];
  float dist = __uint_as_float((unsigned)(key >> 32));
  int idx = (int)(unsigned)key;
  bmuI[b] = (float)loc[2 * idx];
  bmuJ[b] = (float)loc[2 * idx + 1];
  atomicAdd(out_loss, dist);
}

// -------- lrᵀ[c,b] in bf16 + colsum[c] = Σ_b lr --------
__global__ __launch_bounds__(256) void k_lr(const int* __restrict__ loc,
                                            const float* __restrict__ bmuI,
                                            const float* __restrict__ bmuJ,
                                            const int* __restrict__ itp,
                                            bf16_t* __restrict__ lrT,
                                            float* __restrict__ colsum) {
  __shared__ float red[256];
  int c = blockIdx.x, t = threadIdx.x;
  float ci = (float)loc[2 * c], cj = (float)loc[2 * c + 1];
  float decay = 1.f - (float)itp[0] / 1000.0f;
  float alpha = 0.3f * decay;
  float sig   = fmaxf(64.0f * decay, 1.f);
  float inv   = 1.f / (sig * sig);
  float s = 0.f;
  for (int i = 0; i < BS / 256; ++i) {
    int b = t + i * 256;
    float di = ci - bmuI[b], dj = cj - bmuJ[b];
    float lr = alpha * __expf(-(di * di + dj * dj) * inv);
    s += lr;
    lrT[(size_t)c * BS + b] = f2bf(lr);
  }
  red[t] = s; __syncthreads();
  for (int off = 128; off > 0; off >>= 1) { if (t < off) red[t] += red[t + off]; __syncthreads(); }
  if (t == 0) colsum[c] = red[0];
}

// -------- GEMM2: acc = lrᵀ @ X ; new_w = w + (acc - colsum*w)/BS --------
// Same 4x2 register blocking, single base pointer per matrix.
__global__ __launch_bounds__(256) void k_gemm2_update(const bf16_t* __restrict__ lrT,
                                                      const bf16_t* __restrict__ xtbf,
                                                      const float*  __restrict__ weights,
                                                      const float*  __restrict__ colsum,
                                                      float* __restrict__ out_w) {
  const int lane  = threadIdx.x & 31;
  const int wv    = threadIdx.x >> 5;
  const int half  = lane >> 4;
  const int tileC = blockIdx.x * 128 + (wv >> 2) * 64;   // neurons
  const int tileD = blockIdx.y * 128 + (wv & 3) * 32;    // dims
  const int koff  = half << 3;

  const bf16_t* pA = lrT  + (size_t)(tileC + (lane & 15)) * BS + koff;
  const bf16_t* pB = xtbf + (size_t)(tileD + (lane & 15)) * BS + koff;

  v8f acc[4][2];
#pragma unroll
  for (int i = 0; i < 4; ++i)
#pragma unroll
    for (int j = 0; j < 2; ++j) acc[i][j] = (v8f){};

  for (int k = 0; k < BS; k += 32) {
    v16bf af[4], bfr[2];
#pragma unroll
    for (int i = 0; i < 4; ++i) af[i] = ldfrag(pA + (size_t)(i * 16) * BS);
#pragma unroll
    for (int j = 0; j < 2; ++j) bfr[j] = ldfrag(pB + (size_t)(j * 16) * BS);
    __builtin_prefetch(pA + 32, 0, 1);
    __builtin_prefetch(pB + 32, 0, 1);
#pragma unroll
    for (int i = 0; i < 4; ++i)
#pragma unroll
      for (int j = 0; j < 2; ++j)
        acc[i][j] = __builtin_amdgcn_wmma_f32_16x16x32_bf16(
            false, af[i], false, bfr[j], (short)0, acc[i][j], false, false);
    pA += 32; pB += 32;
  }

#pragma unroll
  for (int i = 0; i < 4; ++i) {
#pragma unroll
    for (int r = 0; r < 8; ++r) {
      int c = tileC + i * 16 + r + (half << 3);
      float cs = colsum[c];
#pragma unroll
      for (int j = 0; j < 2; ++j) {
        int d = tileD + j * 16 + (lane & 15);
        float w = weights[(size_t)c * DIM + d];
        out_w[(size_t)c * DIM + d] = w + (acc[i][j][r] - cs * w) * (1.f / (float)BS);
      }
    }
  }
}

extern "C" void kernel_launch(void* const* d_in, const int* in_sizes, int n_in,
                              void* d_out, int out_size, void* d_ws, size_t ws_size,
                              hipStream_t stream) {
  (void)in_sizes; (void)n_in; (void)out_size; (void)ws_size;
  const float* x   = (const float*)d_in[0];   // [BS, DIM]
  const float* w   = (const float*)d_in[1];   // [COMPS, DIM]
  const int*   loc = (const int*)d_in[2];     // [COMPS, 2]
  const int*   itp = (const int*)d_in[3];     // scalar
  float* out = (float*)d_out;                 // [0]=loss, [1..]=new_weights

  char* ws = (char*)d_ws;
  size_t o = 0;
  auto take = [&](size_t bytes) -> void* {
    void* p = ws + o;
    o = (o + bytes + 255) & ~(size_t)255;
    return p;
  };
  unsigned long long* keys = (unsigned long long*)take(BS * 8);
  float*  x2     = (float*)take(BS * 4);
  float*  w2     = (float*)take(COMPS * 4);
  float*  bmuI   = (float*)take(BS * 4);
  float*  bmuJ   = (float*)take(BS * 4);
  float*  colsum = (float*)take(COMPS * 4);
  bf16_t* xbf    = (bf16_t*)take((size_t)BS * DIM * 2);
  bf16_t* xtbf   = (bf16_t*)take((size_t)DIM * BS * 2);
  bf16_t* wbf    = (bf16_t*)take((size_t)COMPS * DIM * 2);
  bf16_t* lrT    = (bf16_t*)take((size_t)COMPS * BS * 2);

  k_init<<<4, 256, 0, stream>>>(keys, out);
  k_conv_x<<<BS, 256, 0, stream>>>(x, x2, xbf, xtbf);
  k_conv_w<<<COMPS, 256, 0, stream>>>(w, w2, wbf);
  k_gemm1_bmu<<<dim3(BS / 128, COMPS / 128), 256, 0, stream>>>(xbf, wbf, x2, w2, keys);
  k_bmu<<<4, 256, 0, stream>>>(keys, loc, bmuI, bmuJ, out);
  k_lr<<<COMPS, 256, 0, stream>>>(loc, bmuI, bmuJ, itp, lrT, colsum);
  k_gemm2_update<<<dim3(COMPS / 128, DIM / 128), 256, 0, stream>>>(lrT, xtbf, w, colsum, out + 1);
}
